// ProbSparseAttention_31997506355456
// MI455X (gfx1250) — matrix operand order repaired and checked by reading, not compiled, not run on hardware
//
#include <hip/hip_runtime.h>
#include <hip/hip_bf16.h>

typedef __bf16 bf16_t;
typedef __attribute__((ext_vector_type(16))) __bf16 v16bf;
typedef __attribute__((ext_vector_type(8)))  __bf16 v8bf;
typedef __attribute__((ext_vector_type(8)))  float  v8f;

#define B_   4
#define L_   2048
#define D_   1024
#define H_   16
#define DK_  64
#define U_   7
#define BH_  (B_*H_)
#define ML_  (B_*L_)          // 8192 rows

// ---- workspace layout (bytes) ----
#define OFF_XB   0ull
#define SZ_XB    ((size_t)ML_*D_*2)        // 16 MB  x in bf16
#define OFF_WQ   (OFF_XB + SZ_XB)
#define SZ_W     ((size_t)D_*D_*2)         // 2 MB each
#define OFF_WK   (OFF_WQ + SZ_W)
#define OFF_WV   (OFF_WK + SZ_W)
#define OFF_Q    (OFF_WV + SZ_W)
#define SZ_QKV   ((size_t)BH_*L_*DK_*2)    // 16 MB each, [B,H,L,64] bf16
#define OFF_K    (OFF_Q + SZ_QKV)
#define OFF_V    (OFF_K + SZ_QKV)
#define OFF_RM   (OFF_V + SZ_QKV)
#define SZ_RM    ((size_t)BH_*L_*4)        // 512 KB rowmax fp32
#define OFF_TOPK (OFF_RM + SZ_RM)
#define SZ_TOPK  (4096ull)
#define OFF_CTX  (OFF_TOPK + SZ_TOPK)      // 448*64 fp32

// LDS K-chunk staging for the score pass
#define CHUNK_KEYS 64
#define ROW_BYTES  160                      // 128B row + 32B pad (bank spread)
#define BUF_BYTES  (CHUNK_KEYS * ROW_BYTES) // 10240 B per buffer

static __device__ __forceinline__ v8f vzero8() {
  v8f z;
#pragma unroll
  for (int i = 0; i < 8; ++i) z[i] = 0.0f;
  return z;
}

// generic -> LDS(addrspace 3) offset for async-DMA destination operand
typedef __attribute__((address_space(3))) char lds_char_t;
static __device__ __forceinline__ unsigned to_lds_off(void* p) {
  return (unsigned)(unsigned long long)(lds_char_t*)(char*)p;
}

static __device__ __forceinline__ void async_ld_b128(unsigned lds_off, const void* gaddr) {
  // ASYNCcnt-tracked DMA: memory -> LDS, no VGPR data path
  asm volatile("global_load_async_to_lds_b128 %0, %1, off"
               :: "v"(lds_off), "v"(gaddr) : "memory");
}

// A-tile (16x32 bf16, MxK) from row-major [*, ld].
// Lanes 0-15: K in {k0..k0+15}; lanes 16-31: K in {k0+16..k0+31}.
static __device__ __forceinline__ v16bf load_a16x32(const bf16_t* __restrict__ base,
                                                    int ld, int k0) {
  int lane = threadIdx.x & 31;
  int m = lane & 15;
  int g = lane >> 4;
  const bf16_t* p = base + (size_t)m * ld + k0 + g * 16;
  v8bf lo = *(const v8bf*)p;
  v8bf hi = *(const v8bf*)(p + 8);
  v16bf r;
#pragma unroll
  for (int i = 0; i < 8; ++i) { r[i] = lo[i]; r[8 + i] = hi[i]; }
  return r;
}

// B-tile (32x16 bf16, KxN): bt row n == B column n, contiguous in K.
static __device__ __forceinline__ v16bf load_bt32x16(const bf16_t* __restrict__ bt,
                                                     int ld, int k0) {
  int lane = threadIdx.x & 31;
  int n = lane & 15;
  int kk = k0 + ((lane >> 4) << 4);
  return *(const v16bf*)(bt + (size_t)n * ld + kk);
}

// B-tile from staged LDS chunk (padded rows)
static __device__ __forceinline__ v16bf lds_bt32x16(const char* buf, int key0, int k0) {
  int lane = threadIdx.x & 31;
  int n = lane & 15;
  int kk = k0 + ((lane >> 4) << 4);               // elements
  return *(const v16bf*)(buf + (size_t)(key0 + n) * ROW_BYTES + kk * 2);
}

// ---------------- kernels ----------------

__global__ void k_f32_to_bf16(const float* __restrict__ in, bf16_t* __restrict__ out, int n) {
  int i = blockIdx.x * blockDim.x + threadIdx.x;
  if (i < n) out[i] = (bf16_t)in[i];
}

// One wave computes a 16x64 tile of X @ W^T + b; K-dim register double-buffered.
__global__ void k_qkv_proj(const bf16_t* __restrict__ xb,
                           const bf16_t* __restrict__ wq, const bf16_t* __restrict__ wk,
                           const bf16_t* __restrict__ wv,
                           const float* __restrict__ bq, const float* __restrict__ bk,
                           const float* __restrict__ bv,
                           bf16_t* __restrict__ Q, bf16_t* __restrict__ K,
                           bf16_t* __restrict__ V) {
  int mt  = blockIdx.x;              // 0..511
  int nb  = blockIdx.y;              // 0..15
  int sel = blockIdx.z;              // 0:Q 1:K 2:V
  const bf16_t* w   = (sel == 0) ? wq : (sel == 1) ? wk : wv;
  const float* bias = (sel == 0) ? bq : (sel == 1) ? bk : bv;
  bf16_t* out       = (sel == 0) ? Q  : (sel == 1) ? K  : V;

  int row0 = mt * 16;
  int col0 = nb * 64;
  const bf16_t* arow = xb + (size_t)row0 * D_;

  v8f acc[4];
#pragma unroll
  for (int j = 0; j < 4; ++j) acc[j] = vzero8();

  // prologue loads
  v16bf a = load_a16x32(arow, D_, 0);
  v16bf b[4];
#pragma unroll
  for (int j = 0; j < 4; ++j)
    b[j] = load_bt32x16(w + (size_t)(col0 + j * 16) * D_, D_, 0);

  for (int k0 = 0; k0 < D_; k0 += 32) {
    v16bf an, bn[4];
    if (k0 + 32 < D_) {                       // prefetch next K-slice
      an = load_a16x32(arow, D_, k0 + 32);
#pragma unroll
      for (int j = 0; j < 4; ++j)
        bn[j] = load_bt32x16(w + (size_t)(col0 + j * 16) * D_, D_, k0 + 32);
    }
#pragma unroll
    for (int j = 0; j < 4; ++j)
      acc[j] = __builtin_amdgcn_wmma_f32_16x16x32_bf16(
          false, a, false, b[j], (short)0, acc[j], false, false);
    a = an;
#pragma unroll
    for (int j = 0; j < 4; ++j) b[j] = bn[j];
  }

  int lane = threadIdx.x & 31;
  int nl = lane & 15;
  int mb = (lane >> 4) * 8;
#pragma unroll
  for (int j = 0; j < 4; ++j) {
    int n = col0 + j * 16 + nl;
    float bi = bias[n];
    int h = n >> 6, d = n & 63;
#pragma unroll
    for (int r = 0; r < 8; ++r) {
      int m = row0 + mb + r;
      int bb = m >> 11, l = m & (L_ - 1);
      size_t idx = (((size_t)bb * H_ + h) * L_ + l) * DK_ + d;
      out[idx] = (bf16_t)(acc[j][r] + bi);
    }
  }
}

// 8 waves / block, 128 queries / block, one (b,h) per block-row.
// K staged into LDS in 64-key chunks via async DMA, double-buffered:
// next chunk's global_load_async_to_lds_b128 overlaps this chunk's 8 WMMAs/wave.
__global__ void k_rowmax(const bf16_t* __restrict__ Q, const bf16_t* __restrict__ K,
                         float* __restrict__ rowmax) {
  __shared__ alignas(32) char kbuf[2][BUF_BYTES];

  int bh = blockIdx.x;               // 0..63
  int qb = blockIdx.y;               // 0..15  (128-query block)
  int tid = threadIdx.x;             // 256
  int wave = tid >> 5;
  int q0 = qb * 128 + wave * 16;

  const bf16_t* qbase = Q + ((size_t)bh * L_ + q0) * DK_;
  const char*   kbase = (const char*)(K + (size_t)bh * L_ * DK_);

  v16bf a0 = load_a16x32(qbase, DK_, 0);
  v16bf a1 = load_a16x32(qbase, DK_, 32);

  float rm[8];
#pragma unroll
  for (int r = 0; r < 8; ++r) rm[r] = -3.4e38f;

  const int NCH = L_ / CHUNK_KEYS;   // 32 chunks of 64 keys (8 KB each)

  // issue chunk 0: 512 x b128 segments, 2 per thread
#pragma unroll
  for (int i = 0; i < 2; ++i) {
    int s = tid + i * 256;
    int row = s >> 3, col = s & 7;
    async_ld_b128(to_lds_off(&kbuf[0][row * ROW_BYTES + col * 16]),
                  kbase + (size_t)s * 16);
  }

  for (int c = 0; c < NCH; ++c) {
    if (c + 1 < NCH) {
      const char* g = kbase + (size_t)(c + 1) * CHUNK_KEYS * DK_ * 2;
      char* dbuf = kbuf[(c + 1) & 1];
#pragma unroll
      for (int i = 0; i < 2; ++i) {
        int s = tid + i * 256;
        int row = s >> 3, col = s & 7;
        async_ld_b128(to_lds_off(&dbuf[row * ROW_BYTES + col * 16]),
                      g + (size_t)s * 16);
      }
      asm volatile("s_wait_asynccnt 0x2" ::: "memory");  // chunk c done, c+1 in flight
    } else {
      asm volatile("s_wait_asynccnt 0x0" ::: "memory");
    }
    __syncthreads();

    const char* sb = kbuf[c & 1];
#pragma unroll
    for (int j = 0; j < 4; ++j) {                 // 4 key tiles of 16
      v16bf b0 = lds_bt32x16(sb, j * 16, 0);
      v16bf b1 = lds_bt32x16(sb, j * 16, 32);
      v8f cc = vzero8();
      cc = __builtin_amdgcn_wmma_f32_16x16x32_bf16(false, a0, false, b0, (short)0, cc, false, false);
      cc = __builtin_amdgcn_wmma_f32_16x16x32_bf16(false, a1, false, b1, (short)0, cc, false, false);
#pragma unroll
      for (int r = 0; r < 8; ++r) rm[r] = fmaxf(rm[r], cc[r]);
    }
    __syncthreads();                              // buffer (c&1) free for chunk c+2
  }

  // reduce across the 16 lanes holding the same rows (wave32: xor stays in half)
#pragma unroll
  for (int r = 0; r < 8; ++r) {
#pragma unroll
    for (int m = 1; m < 16; m <<= 1) rm[r] = fmaxf(rm[r], __shfl_xor(rm[r], m, 32));
  }
  int lane = tid & 31;
  if ((lane & 15) == 0) {
    int mb = (lane >> 4) * 8;
    const float scale = 0.125f;      // 1/sqrt(64)
#pragma unroll
    for (int r = 0; r < 8; ++r)
      rowmax[(size_t)bh * L_ + q0 + mb + r] = rm[r] * scale;
  }
}

// Top-7 argmax per (b,h) via 7 rounds of block-wide argmax.
__global__ void k_topk(const float* __restrict__ rowmax, int* __restrict__ topk) {
  __shared__ float sv[256];
  __shared__ int   si[256];
  int bh = blockIdx.x;
  int t = threadIdx.x;
  const float* rm = rowmax + (size_t)bh * L_;
  int chosen[U_];
  for (int it = 0; it < U_; ++it) {
    float bestv = -3.4e38f; int besti = 0x7fffffff;
    for (int i = t; i < L_; i += 256) {
      bool skip = false;
      for (int j = 0; j < it; ++j) if (chosen[j] == i) skip = true;
      float v = rm[i];
      if (!skip && (v > bestv || (v == bestv && i < besti))) { bestv = v; besti = i; }
    }
    sv[t] = bestv; si[t] = besti;
    __syncthreads();
    for (int s = 128; s > 0; s >>= 1) {
      if (t < s) {
        if (sv[t + s] > sv[t] || (sv[t + s] == sv[t] && si[t + s] < si[t])) {
          sv[t] = sv[t + s]; si[t] = si[t + s];
        }
      }
      __syncthreads();
    }
    int win = si[0];
    chosen[it] = win;
    if (t == 0) topk[bh * U_ + it] = win;
    __syncthreads();
  }
}

// Per (b,h): softmax over 7 selected queries x all keys, ctx = attn @ V.
__global__ void k_sparse_attn(const bf16_t* __restrict__ Q, const bf16_t* __restrict__ K,
                              const bf16_t* __restrict__ V, const int* __restrict__ topk,
                              float* __restrict__ ctx) {
  __shared__ float sc[U_][L_];     // 56 KB of the 320 KB WGP LDS
  __shared__ float qs[U_][DK_];
  __shared__ int   sel[U_];
  int bh = blockIdx.x;
  int t = threadIdx.x;
  if (t < U_) sel[t] = topk[bh * U_ + t];
  __syncthreads();
  if (t < U_ * DK_) {
    int u = t >> 6, d = t & 63;
    qs[u][d] = (float)Q[((size_t)bh * L_ + sel[u]) * DK_ + d];
  }
  __syncthreads();

  const float scale = 0.125f;
  const bf16_t* kb = K + (size_t)bh * L_ * DK_;
  for (int k = t; k < L_; k += 256) {
    float acc[U_];
#pragma unroll
    for (int u = 0; u < U_; ++u) acc[u] = 0.0f;
    const bf16_t* kr = kb + (size_t)k * DK_;
    for (int d = 0; d < DK_; ++d) {
      float kv = (float)kr[d];
#pragma unroll
      for (int u = 0; u < U_; ++u) acc[u] += qs[u][d] * kv;
    }
#pragma unroll
    for (int u = 0; u < U_; ++u) sc[u][k] = acc[u] * scale;
  }
  __syncthreads();

  int w = t >> 5, lane = t & 31;
  if (w < U_) {
    float mx = -3.4e38f;
    for (int k = lane; k < L_; k += 32) mx = fmaxf(mx, sc[w][k]);
#pragma unroll
    for (int m = 1; m < 32; m <<= 1) mx = fmaxf(mx, __shfl_xor(mx, m, 32));
    float sum = 0.0f;
    for (int k = lane; k < L_; k += 32) { float e = __expf(sc[w][k] - mx); sc[w][k] = e; sum += e; }
#pragma unroll
    for (int m = 1; m < 32; m <<= 1) sum += __shfl_xor(sum, m, 32);
    float inv = 1.0f / sum;
    for (int k = lane; k < L_; k += 32) sc[w][k] *= inv;
  }
  __syncthreads();

  if (t < U_ * DK_) {
    int u = t >> 6, d = t & 63;
    const bf16_t* vb = V + (size_t)bh * L_ * DK_ + d;
    float acc = 0.0f;
    for (int k = 0; k < L_; ++k) acc += sc[u][k] * (float)vb[(size_t)k * DK_];
    ctx[((size_t)bh * U_ + u) * DK_ + d] = acc;
  }
}

__global__ void k_bias_fill(const float* __restrict__ bo, float* __restrict__ out, int n) {
  int i = blockIdx.x * blockDim.x + threadIdx.x;
  if (i < n) out[i] = bo[i & (D_ - 1)];
}

// One block per (b,h,u): out[b,l,:] += ctx_row @ wo[:, h*64:(h+1)*64]^T
__global__ void k_out_scatter(const float* __restrict__ ctx, const int* __restrict__ topk,
                              const float* __restrict__ wo, float* __restrict__ out) {
  __shared__ float cvec[DK_];
  __shared__ int lsel;
  int blk = blockIdx.x;            // 0..447
  int u = blk % U_;
  int bh = blk / U_;
  int b = bh >> 4, h = bh & 15;
  int t = threadIdx.x;
  if (t == 0) lsel = topk[bh * U_ + u];
  if (t < DK_) cvec[t] = ctx[((size_t)bh * U_ + u) * DK_ + t];
  __syncthreads();
  int l = lsel;
  float* orow = out + ((size_t)b * L_ + l) * D_;
  for (int j = t; j < D_; j += 256) {
    const float* wrow = wo + (size_t)j * D_ + h * DK_;
    float s = 0.0f;
#pragma unroll
    for (int d = 0; d < DK_; ++d) s += cvec[d] * wrow[d];
    atomicAdd(&orow[j], s);
  }
}

// ---------------- launch ----------------

extern "C" void kernel_launch(void* const* d_in, const int* in_sizes, int n_in,
                              void* d_out, int out_size, void* d_ws, size_t ws_size,
                              hipStream_t stream) {
  const float* x  = (const float*)d_in[0];
  const float* wq = (const float*)d_in[1];
  const float* bq = (const float*)d_in[2];
  const float* wk = (const float*)d_in[3];
  const float* bk = (const float*)d_in[4];
  const float* wv = (const float*)d_in[5];
  const float* bv = (const float*)d_in[6];
  const float* wo = (const float*)d_in[7];
  const float* bo = (const float*)d_in[8];
  float* out = (float*)d_out;

  char* ws = (char*)d_ws;
  bf16_t* xb  = (bf16_t*)(ws + OFF_XB);
  bf16_t* wqb = (bf16_t*)(ws + OFF_WQ);
  bf16_t* wkb = (bf16_t*)(ws + OFF_WK);
  bf16_t* wvb = (bf16_t*)(ws + OFF_WV);
  bf16_t* Qb  = (bf16_t*)(ws + OFF_Q);
  bf16_t* Kb  = (bf16_t*)(ws + OFF_K);
  bf16_t* Vb  = (bf16_t*)(ws + OFF_V);
  float*  rmx = (float*)(ws + OFF_RM);
  int*    tki = (int*)(ws + OFF_TOPK);
  float*  ctx = (float*)(ws + OFF_CTX);

  k_f32_to_bf16<<<(ML_ * D_ + 255) / 256, 256, 0, stream>>>(x, xb, ML_ * D_);
  k_f32_to_bf16<<<(D_ * D_ + 255) / 256, 256, 0, stream>>>(wq, wqb, D_ * D_);
  k_f32_to_bf16<<<(D_ * D_ + 255) / 256, 256, 0, stream>>>(wk, wkb, D_ * D_);
  k_f32_to_bf16<<<(D_ * D_ + 255) / 256, 256, 0, stream>>>(wv, wvb, D_ * D_);

  k_qkv_proj<<<dim3(ML_ / 16, D_ / 64, 3), 32, 0, stream>>>(
      xb, wqb, wkb, wvb, bq, bk, bv, Qb, Kb, Vb);

  k_rowmax<<<dim3(BH_, L_ / 128), 256, 0, stream>>>(Qb, Kb, rmx);

  k_topk<<<BH_, 256, 0, stream>>>(rmx, tki);

  k_sparse_attn<<<BH_, 256, 0, stream>>>(Qb, Kb, Vb, tki, ctx);

  k_bias_fill<<<(ML_ * D_ + 255) / 256, 256, 0, stream>>>(bo, out, ML_ * D_);
  k_out_scatter<<<BH_ * U_, 256, 0, stream>>>(ctx, tki, wo, out);
}
